// Encoder_77704548319663
// MI455X (gfx1250) — matrix operand order repaired and checked by reading, not compiled, not run on hardware
//
#include <hip/hip_runtime.h>
#include <hip/hip_bf16.h>
#include <math.h>

// ---------------------------------------------------------------------------
// R-NET forward for MI455X (gfx1250, wave32, WMMA 16x16x32 f16 -> f32 accum)
// Fragment traffic fully vectorized: A frags = 2x b128 loads, B frags packed
// fragment-major at cast time = 2x b128 loads, attention frags = 4x float4.
//
// Assumed d_in flatten order = jax.tree_util leaves (dict keys sorted):
//   0 passage[1,1000] i32, 1 question[1,100] i32, then params alphabetical
//   (see index map in kernel_launch).
// d_out: start log-softmax [1000] then end [1000], float32.
// ---------------------------------------------------------------------------

typedef _Float16 h16;
typedef __attribute__((ext_vector_type(16))) _Float16 v16h;
typedef __attribute__((ext_vector_type(8)))  _Float16 v8h;
typedef __attribute__((ext_vector_type(8)))  float    v8f;
typedef __attribute__((ext_vector_type(4)))  float    f4;

#define H_    75
#define H3    225
#define P_LEN 1000
#define Q_LEN 100
#define E_DIM 300
// padded dims (K multiples of 32 for WMMA, N multiples of 16)
#define KE   320   // E 300 -> 320
#define K2H  160   // 2H 150 -> 160
#define KH   96    // H 75 -> 96 (also ld for qproj/pproj so frag loads need no guards)
#define N3H  240   // 3H 225 -> 240
#define NH   80    // H 75 -> 80

__device__ __forceinline__ float sigm_(float x) { return 1.f / (1.f + __expf(-x)); }

// ---- WMMA fragment loaders (layouts per CDNA5 ISA 7.12.2, wave32) ----------
// A 16x32 f16: element j -> m = lane&15 ; k = (j&7) + 16*(j>>3) + 8*(lane>>4)
//   => per lane: halves [k0+8g .. +7] and [k0+16+8g .. +7]  (2x 16B loads)
// B 32x16 f16: element j -> k = (j&7) + 16*(j>>3) + 8*(lane>>4) ; n = lane&15 (packed at cast)
// C 16x16 f32: element r -> m = r + 8*(lane>>4) ; n = lane&15

__device__ __forceinline__ v16h a_frag_f16(const h16* __restrict__ A, int lda,
                                           int mt, int k0, int lane) {
  int m = mt * 16 + (lane & 15), g = lane >> 4;
  const h16* p = A + (size_t)m * lda + k0 + 8 * g;
  v8h lo = *(const v8h*)p;
  v8h hi = *(const v8h*)(p + 16);
  v16h a;
#pragma unroll
  for (int j = 0; j < 8; ++j) { a[j] = lo[j]; a[j + 8] = hi[j]; }
  return a;
}

// B packed fragment-major: dst[(kb*Nt + nt)*512 + lane*16 + j]
__device__ __forceinline__ v16h b_frag_pk(const h16* __restrict__ Bpk, int Nt,
                                          int kb, int nt, int lane) {
  const h16* p = Bpk + ((size_t)(kb * Nt + nt) << 9) + lane * 16;
  v8h lo = *(const v8h*)p;
  v8h hi = *(const v8h*)(p + 8);
  v16h b;
#pragma unroll
  for (int j = 0; j < 8; ++j) { b[j] = lo[j]; b[j + 8] = hi[j]; }
  return b;
}

// A fragment computed on the fly: A[q][k] = tanh(proj[q][k] + s[k]).
// proj has ld = KH(96) with cols 75..95 == 0 and sS[75..95] == 0, and rows
// beyond the valid M are zero (softmax consumers mask them), so no guards.
__device__ __forceinline__ v16h attn_a_frag(const float* __restrict__ proj,
                                            const float* sS, int mt, int k0,
                                            int lane) {
  int m = mt * 16 + (lane & 15), g = lane >> 4;
  const float* p = proj + (size_t)m * KH + k0 + 8 * g;
  f4 x0 = *(const f4*)(p);
  f4 x1 = *(const f4*)(p + 4);
  f4 x2 = *(const f4*)(p + 16);
  f4 x3 = *(const f4*)(p + 20);
  int kb = k0 + 8 * g;
  v16h a;
#pragma unroll
  for (int j = 0; j < 4; ++j) {
    a[j]      = (h16)tanhf(x0[j] + sS[kb + j]);
    a[j + 4]  = (h16)tanhf(x1[j] + sS[kb + 4 + j]);
    a[j + 8]  = (h16)tanhf(x2[j] + sS[kb + 16 + j]);
    a[j + 12] = (h16)tanhf(x3[j] + sS[kb + 20 + j]);
  }
  return a;
}

// ---- generic WMMA GEMM: C[M,Nvalid] = A[M,Kp](f16) @ Bpk + bias ------------
__global__ void __launch_bounds__(128)
wmma_gemm_bias(const h16* __restrict__ A, int lda,
               const h16* __restrict__ Bpk, int Nt,
               const float* __restrict__ bias,
               float* __restrict__ C, int ldc,
               int M, int Nvalid, int Kp, int Mt) {
  int tile = blockIdx.x * 4 + (threadIdx.x >> 5);
  if (tile >= Mt * Nt) return;                       // wave-uniform
  int mt = tile % Mt, nt = tile / Mt;
  int lane = threadIdx.x & 31, g = lane >> 4, nl = lane & 15;
  v8f acc = {};
  int kb = 0;
  for (int k0 = 0; k0 < Kp; k0 += 32, ++kb) {
    v16h a = a_frag_f16(A, lda, mt, k0, lane);
    v16h b = b_frag_pk(Bpk, Nt, kb, nt, lane);
    acc = __builtin_amdgcn_wmma_f32_16x16x32_f16(false, a, false, b,
                                                 (short)0, acc, false, false);
  }
  int n = nt * 16 + nl;
#pragma unroll
  for (int r = 0; r < 8; ++r) {
    int m = mt * 16 + r + 8 * g;
    if (m < M && n < Nvalid) C[(size_t)m * ldc + n] = acc[r] + bias[n];
  }
}

// ---- weight pack: fragment-major f16 B from W[N,K] f32 ---------------------
// dst[(kb*Nt + nt)*512 + lane*16 + j] = W[n,k], k = kb*32 + (j&7) + 8*(lane>>4) + 16*(j>>3),
// n = nt*16 + (lane&15); zero-padded.
__global__ void pack_wB_kernel(h16* __restrict__ dst, const float* __restrict__ src,
                               int N, int K, int Np, int Kp) {
  int i = blockIdx.x * blockDim.x + threadIdx.x;
  if (i >= Kp * Np) return;
  int j = i & 15;
  int lane = (i >> 4) & 31;
  int block = i >> 9;
  int Nt = Np / 16;
  int kb = block / Nt, nt = block % Nt;
  int k = kb * 32 + (j & 7) + ((lane >> 4) << 3) + ((j >> 3) << 4);
  int n = nt * 16 + (lane & 15);
  dst[i] = (k < K && n < N) ? (h16)src[(size_t)n * K + k] : (h16)0.f;
}

// ---- embedding gather -> f16 padded [T,KE] ---------------------------------
__global__ void gather_emb_kernel(h16* __restrict__ dst, const int* __restrict__ tok,
                                  const float* __restrict__ emb, int T) {
  int i = blockIdx.x * blockDim.x + threadIdx.x;
  if (i >= T * KE) return;
  int r = i / KE, k = i % KE;
  dst[i] = (k < E_DIM) ? (h16)emb[(size_t)tok[r] * E_DIM + k] : (h16)0.f;
}

// ---- sequential GRU scan (persistent, 1 block, 256 threads) ----------------
__global__ void __launch_bounds__(256)
gru_scan_kernel(const float* __restrict__ gi, int ldgi,
                const float* __restrict__ Whh, const float* __restrict__ bhh,
                float* __restrict__ y, h16* __restrict__ yh, int ldy, int yoff,
                int T, int reverse) {
  __shared__ float hS[80];
  __shared__ float ghS[228];
  int tid = threadIdx.x;
  if (tid < 80) hS[tid] = 0.f;
  __syncthreads();
  for (int t = 0; t < T; ++t) {
    int idx = reverse ? (T - 1 - t) : t;
    if (tid < H3) {
      float s = bhh[tid];
      const float* w = Whh + tid * H_;
      for (int k = 0; k < H_; ++k) s += w[k] * hS[k];
      ghS[tid] = s;
    }
    __syncthreads();
    if (tid < H_) {
      const float* gr = gi + (size_t)idx * ldgi;
      float r = sigm_(gr[tid] + ghS[tid]);
      float z = sigm_(gr[H_ + tid] + ghS[H_ + tid]);
      float n = tanhf(gr[2 * H_ + tid] + r * ghS[2 * H_ + tid]);
      float hn = (1.f - z) * n + z * hS[tid];
      hS[tid] = hn;
      y[(size_t)idx * ldy + yoff + tid] = hn;
      yh[(size_t)idx * ldy + yoff + tid] = (h16)hn;
    }
    __syncthreads();
  }
}

// ---- attention-pooled question vector r_q ----------------------------------
__global__ void pool_question_kernel(const float* __restrict__ qproj,
                                     const float* __restrict__ vtW,
                                     const float* __restrict__ vtb,
                                     float* __restrict__ rq) {
  int h = threadIdx.x;
  if (h >= H_) return;
  float acc = 0.f;
  for (int q = 0; q < Q_LEN; ++q) {
    const float* row = qproj + q * KH;
    float s = vtb[h];
    for (int k = 0; k < H_; ++k) s += vtW[h * H_ + k] * tanhf(row[k]);
    acc += sigm_(s) * row[h];
  }
  rq[h] = acc;
}

// ---- shared gated bi-GRU step (fi -> gate -> GRU f/b), LDS resident --------
__device__ void gated_bigru_step(int tid,
    const float* gateW, const float* gateB,
    const float* WihF, const float* bihF, const float* WhhF, const float* bhhF,
    const float* WihB, const float* bihB, const float* WhhB, const float* bhhB,
    float* fiS, float* gateS, float* gifS, float* gibS, float* ghfS, float* ghbS,
    float* hfS, float* hbS) {
  if (tid < H3) {
    float s = gateB[tid];
    const float* w = gateW + tid * H3;
    for (int k = 0; k < H3; ++k) s += w[k] * fiS[k];
    gateS[tid] = sigm_(s);
  }
  __syncthreads();
  if (tid < H3) fiS[tid] *= gateS[tid];
  __syncthreads();
  if (tid < H3) {
    float sf = bihF[tid], sb = bihB[tid];
    const float* wf = WihF + tid * H3;
    const float* wb = WihB + tid * H3;
    for (int k = 0; k < H3; ++k) { float x = fiS[k]; sf += wf[k] * x; sb += wb[k] * x; }
    gifS[tid] = sf; gibS[tid] = sb;
    float gf = bhhF[tid], gb = bhhB[tid];
    const float* hf = WhhF + tid * H_;
    const float* hb = WhhB + tid * H_;
    for (int k = 0; k < H_; ++k) { gf += hf[k] * hfS[k]; gb += hb[k] * hbS[k]; }
    ghfS[tid] = gf; ghbS[tid] = gb;
  }
  __syncthreads();
  if (tid < H_) {
    float r = sigm_(gifS[tid] + ghfS[tid]);
    float z = sigm_(gifS[H_ + tid] + ghfS[H_ + tid]);
    float n = tanhf(gifS[2 * H_ + tid] + r * ghfS[2 * H_ + tid]);
    hfS[tid] = (1.f - z) * n + z * hfS[tid];
    r = sigm_(gibS[tid] + ghbS[tid]);
    z = sigm_(gibS[H_ + tid] + ghbS[H_ + tid]);
    n = tanhf(gibS[2 * H_ + tid] + r * ghbS[2 * H_ + tid]);
    hbS[tid] = (1.f - z) * n + z * hbS[tid];
  }
  __syncthreads();
}

// ---- question-aware passage scan (WMMA attention each step) ----------------
__global__ void __launch_bounds__(512)
qa_scan_kernel(const float* __restrict__ qproj, const float* __restrict__ pprA,
               const float* __restrict__ uP, const h16* __restrict__ vtPk,
               const float* linHidW, const float* linHidB,
               const float* gateW, const float* gateB,
               const float* WihF, const float* WhhF, const float* bihF, const float* bhhF,
               const float* WihB, const float* WhhB, const float* bihB, const float* bhhB,
               float* __restrict__ v, h16* __restrict__ vh, int T) {
  __shared__ float scS[112][NH];          // attention scores, Q padded to 112
  __shared__ float sS[KH];
  __shared__ float cS[NH], hfS[NH], hbS[NH];
  __shared__ float fiS[240], gateS[240], gifS[240], gibS[240], ghfS[240], ghbS[240];
  int tid = threadIdx.x, lane = tid & 31, w = tid >> 5;
  if (tid < NH) { hfS[tid] = 0.f; hbS[tid] = 0.f; }
  __syncthreads();
  for (int t = 0; t < T; ++t) {
    // s = lin_hidden([hf,hb]) + ppr_all[t]  (zero for k >= 75)
    if (tid < KH) {
      float s = 0.f;
      if (tid < H_) {
        float acc = linHidB[tid];
        const float* wr = linHidW + tid * 2 * H_;
        for (int k = 0; k < H_; ++k) acc += wr[k] * hfS[k] + wr[H_ + k] * hbS[k];
        s = acc + pprA[(size_t)t * NH + tid];
      }
      sS[tid] = s;
    }
    __syncthreads();
    // sc = tanh(qproj + s) @ vtW^T  via WMMA: 7 x 5 tiles over 16 waves
    for (int tile = w; tile < 35; tile += 16) {
      int mt = tile % 7, nt = tile / 7;
      v8f acc = {};
#pragma unroll
      for (int kt = 0; kt < 3; ++kt) {
        v16h a = attn_a_frag(qproj, sS, mt, kt * 32, lane);
        v16h b = b_frag_pk(vtPk, NH / 16, kt, nt, lane);
        acc = __builtin_amdgcn_wmma_f32_16x16x32_f16(false, a, false, b,
                                                     (short)0, acc, false, false);
      }
      int n = nt * 16 + (lane & 15), g = lane >> 4;
#pragma unroll
      for (int r = 0; r < 8; ++r) scS[mt * 16 + r + 8 * g][n] = acc[r];
    }
    __syncthreads();
    // per-column softmax over Q, weighted sum with qproj
    if (tid < H_) {
      float m = -1e30f;
      for (int q = 0; q < Q_LEN; ++q) m = fmaxf(m, scS[q][tid]);
      float s = 0.f, a = 0.f;
      for (int q = 0; q < Q_LEN; ++q) {
        float e = __expf(scS[q][tid] - m);
        s += e; a += e * qproj[(size_t)q * KH + tid];
      }
      cS[tid] = a / s;
    }
    __syncthreads();
    if (tid < 2 * H_) fiS[tid] = uP[(size_t)t * K2H + tid];
    else if (tid < H3) fiS[tid] = cS[tid - 2 * H_];
    __syncthreads();
    gated_bigru_step(tid, gateW, gateB, WihF, bihF, WhhF, bhhF,
                     WihB, bihB, WhhB, bhhB,
                     fiS, gateS, gifS, gibS, ghfS, ghbS, hfS, hbS);
    if (tid < H_) {
      float hf = hfS[tid], hb = hbS[tid];
      v[(size_t)t * K2H + tid] = hf;       v[(size_t)t * K2H + H_ + tid] = hb;
      vh[(size_t)t * K2H + tid] = (h16)hf; vh[(size_t)t * K2H + H_ + tid] = (h16)hb;
    }
    __syncthreads();
  }
}

// ---- self-matching scan (WMMA + online softmax over P=1000) ----------------
__global__ void __launch_bounds__(512)
sm_scan_kernel(const float* __restrict__ pproj, const float* __restrict__ pwrA,
               const float* __restrict__ v, const h16* __restrict__ vtPk,
               const float* gateW, const float* gateB,
               const float* WihF, const float* WhhF, const float* bihF, const float* bhhF,
               const float* WihB, const float* WhhB, const float* bihB, const float* bhhB,
               float* __restrict__ hsm, h16* __restrict__ hsmh, int T) {
  __shared__ float scS[48][NH];           // one 3-mtile chunk of scores
  __shared__ float maxS[NH], sumS[NH], accS[NH];
  __shared__ float sS[KH];
  __shared__ float cS[NH], hfS[NH], hbS[NH];
  __shared__ float fiS[240], gateS[240], gifS[240], gibS[240], ghfS[240], ghbS[240];
  int tid = threadIdx.x, lane = tid & 31, w = tid >> 5;
  if (tid < NH) { hfS[tid] = 0.f; hbS[tid] = 0.f; }
  __syncthreads();
  for (int t = 0; t < T; ++t) {
    if (tid < KH) sS[tid] = (tid < H_) ? pwrA[(size_t)t * NH + tid] : 0.f;
    if (tid < NH) { maxS[tid] = -1e30f; sumS[tid] = 0.f; accS[tid] = 0.f; }
    __syncthreads();
    // 63 m-tiles x 5 n-tiles; 3 m-tiles (48 rows) per round, online softmax
    for (int c0 = 0; c0 < 63; c0 += 3) {
      if (w < 15) {
        int mt = c0 + w / 5, nt = w % 5;
        v8f acc = {};
#pragma unroll
        for (int kt = 0; kt < 3; ++kt) {
          v16h a = attn_a_frag(pproj, sS, mt, kt * 32, lane);
          v16h b = b_frag_pk(vtPk, NH / 16, kt, nt, lane);
          acc = __builtin_amdgcn_wmma_f32_16x16x32_f16(false, a, false, b,
                                                       (short)0, acc, false, false);
        }
        int n = nt * 16 + (lane & 15), g = lane >> 4;
#pragma unroll
        for (int r = 0; r < 8; ++r) scS[(mt - c0) * 16 + r + 8 * g][n] = acc[r];
      }
      __syncthreads();
      if (tid < H_) {
        float mloc = -1e30f;
        for (int lq = 0; lq < 48; ++lq)
          if (c0 * 16 + lq < P_LEN) mloc = fmaxf(mloc, scS[lq][tid]);
        float mnew = fmaxf(maxS[tid], mloc);
        float scale = __expf(maxS[tid] - mnew);
        float s = sumS[tid] * scale, a = accS[tid] * scale;
        for (int lq = 0; lq < 48; ++lq) {
          int q = c0 * 16 + lq;
          if (q < P_LEN) {
            float e = __expf(scS[lq][tid] - mnew);
            s += e; a += e * pproj[(size_t)q * KH + tid];
          }
        }
        maxS[tid] = mnew; sumS[tid] = s; accS[tid] = a;
      }
      __syncthreads();
    }
    if (tid < H_) cS[tid] = accS[tid] / sumS[tid];
    __syncthreads();
    if (tid < 2 * H_) fiS[tid] = v[(size_t)t * K2H + tid];
    else if (tid < H3) fiS[tid] = cS[tid - 2 * H_];
    __syncthreads();
    gated_bigru_step(tid, gateW, gateB, WihF, bihF, WhhF, bhhF,
                     WihB, bihB, WhhB, bhhB,
                     fiS, gateS, gifS, gibS, ghfS, ghbS, hfS, hbS);
    if (tid < H_) {
      float hf = hfS[tid], hb = hbS[tid];
      hsm[(size_t)t * K2H + tid] = hf;       hsm[(size_t)t * K2H + H_ + tid] = hb;
      hsmh[(size_t)t * K2H + tid] = (h16)hf; hsmh[(size_t)t * K2H + H_ + tid] = (h16)hb;
    }
    __syncthreads();
  }
}

// ---- pointer network (2 steps) ---------------------------------------------
__global__ void __launch_bounds__(256)
pointer_kernel(const float* __restrict__ hp, const float* __restrict__ rq,
               const float* haW, const float* haB,
               const float* vaW, const float* vaB,
               const float* ansWih, const float* ansWhh,
               const float* ansBih, const float* ansBhh,
               float* __restrict__ out) {
  __shared__ float aS[P_LEN];
  __shared__ float haS[NH], hidS[NH], cS[NH];
  __shared__ float giS[H3], ghS[H3];
  __shared__ float red[2];
  int tid = threadIdx.x;
  if (tid < NH) hidS[tid] = (tid < H_) ? rq[tid] : 0.f;
  __syncthreads();
  for (int iter = 0; iter < 2; ++iter) {
    if (tid < H_) {
      float s = haB[tid];
      for (int k = 0; k < H_; ++k) s += haW[tid * H_ + k] * hidS[k];
      haS[tid] = s;
    }
    __syncthreads();
    for (int q = tid; q < P_LEN; q += 256) {
      float s = vaB[0];
      for (int k = 0; k < H_; ++k) s += vaW[k] * tanhf(hp[(size_t)q * NH + k] + haS[k]);
      aS[q] = s;
    }
    __syncthreads();
    if (tid == 0) {
      float m = -1e30f;
      for (int q = 0; q < P_LEN; ++q) m = fmaxf(m, aS[q]);
      float s = 0.f;
      for (int q = 0; q < P_LEN; ++q) s += __expf(aS[q] - m);
      red[0] = m; red[1] = logf(s);
    }
    __syncthreads();
    for (int q = tid; q < P_LEN; q += 256)
      out[iter * P_LEN + q] = aS[q] - red[0] - red[1];
    if (tid < H_) {
      float a = 0.f;                        // pre-softmax weights, as in source
      for (int q = 0; q < P_LEN; ++q) a += aS[q] * hp[(size_t)q * NH + tid];
      cS[tid] = a;
    }
    __syncthreads();
    if (tid < H3) {
      float gi = ansBih[tid], gh = ansBhh[tid];
      for (int k = 0; k < H_; ++k) {
        gi += ansWih[tid * H_ + k] * cS[k];
        gh += ansWhh[tid * H_ + k] * hidS[k];
      }
      giS[tid] = gi; ghS[tid] = gh;
    }
    __syncthreads();
    if (tid < H_) {
      float r = sigm_(giS[tid] + ghS[tid]);
      float z = sigm_(giS[H_ + tid] + ghS[H_ + tid]);
      float n = tanhf(giS[2 * H_ + tid] + r * ghS[2 * H_ + tid]);
      hidS[tid] = (1.f - z) * n + z * hidS[tid];
    }
    __syncthreads();
  }
}

// ---------------------------------------------------------------------------
extern "C" void kernel_launch(void* const* d_in, const int* in_sizes, int n_in,
                              void* d_out, int out_size, void* d_ws, size_t ws_size,
                              hipStream_t stream) {
  (void)in_sizes; (void)n_in; (void)out_size;
#define INF(i) ((const float*)d_in[(i)])
  const int* passage  = (const int*)d_in[0];
  const int* question = (const int*)d_in[1];
  const float* emb = INF(6);
  const float* encWhhB[3] = { INF(7),  INF(15), INF(23) };
  const float* encWihB[3] = { INF(8),  INF(16), INF(24) };
  const float* encBhhB[3] = { INF(9),  INF(17), INF(25) };
  const float* encBihB[3] = { INF(10), INF(18), INF(26) };
  const float* encWhhF[3] = { INF(11), INF(19), INF(27) };
  const float* encWihF[3] = { INF(12), INF(20), INF(28) };
  const float* encBhhF[3] = { INF(13), INF(21), INF(29) };
  const float* encBihF[3] = { INF(14), INF(22), INF(30) };

  // workspace bump allocator
  size_t off = 0;
  auto alloc = [&](size_t bytes) -> void* {
    void* p = (char*)d_ws + off;
    off += (bytes + 255) & ~(size_t)255;
    return p;
  };
  h16*  xPh   = (h16*)  alloc((size_t)1008 * KE * sizeof(h16));
  h16*  xQh   = (h16*)  alloc((size_t)112  * KE * sizeof(h16));
  h16*  x1h   = (h16*)  alloc((size_t)1008 * K2H * sizeof(h16));
  h16*  x2h   = (h16*)  alloc((size_t)1008 * K2H * sizeof(h16));
  float* giF  = (float*)alloc((size_t)1008 * N3H * sizeof(float));
  float* giB  = (float*)alloc((size_t)1008 * N3H * sizeof(float));
  float* uP   = (float*)alloc((size_t)1008 * K2H * sizeof(float));
  h16*  uPh   = (h16*)  alloc((size_t)1008 * K2H * sizeof(h16));
  float* uQ   = (float*)alloc((size_t)112  * K2H * sizeof(float));
  h16*  uQh   = (h16*)  alloc((size_t)112  * K2H * sizeof(h16));
  float* qproj= (float*)alloc((size_t)112  * KH * sizeof(float));   // ld 96
  float* pprA = (float*)alloc((size_t)1008 * NH * sizeof(float));
  float* vbuf = (float*)alloc((size_t)1008 * K2H * sizeof(float));
  h16*  vh    = (h16*)  alloc((size_t)1008 * K2H * sizeof(h16));
  float* pproj= (float*)alloc((size_t)1008 * KH * sizeof(float));   // ld 96
  float* pwrA = (float*)alloc((size_t)1008 * NH * sizeof(float));
  float* hsm  = (float*)alloc((size_t)1008 * K2H * sizeof(float));
  h16*  hsmh  = (h16*)  alloc((size_t)1008 * K2H * sizeof(h16));
  float* hp   = (float*)alloc((size_t)1008 * NH * sizeof(float));
  float* rq   = (float*)alloc((size_t)NH * sizeof(float));
  h16* wEncFt[3]; h16* wEncBt[3];
  for (int l = 0; l < 3; ++l) {
    int Kp = (l == 0) ? KE : K2H;
    wEncFt[l] = (h16*)alloc((size_t)Kp * N3H * sizeof(h16));
    wEncBt[l] = (h16*)alloc((size_t)Kp * N3H * sizeof(h16));
  }
  h16* linQt   = (h16*)alloc((size_t)K2H * NH * sizeof(h16));
  h16* linPt   = (h16*)alloc((size_t)K2H * NH * sizeof(h16));
  h16* linSPt  = (h16*)alloc((size_t)K2H * NH * sizeof(h16));
  h16* linSPWt = (h16*)alloc((size_t)K2H * NH * sizeof(h16));
  h16* linHPt  = (h16*)alloc((size_t)K2H * NH * sizeof(h16));
  h16* vtPk    = (h16*)alloc((size_t)KH * NH * sizeof(h16));

  hipMemsetAsync(d_ws, 0, off < ws_size ? off : ws_size, stream);

  auto pack = [&](h16* dst, const float* W, int N, int K, int Np, int Kp) {
    int tot = Kp * Np;
    pack_wB_kernel<<<(tot + 255) / 256, 256, 0, stream>>>(dst, W, N, K, Np, Kp);
  };
  for (int l = 0; l < 3; ++l) {
    int K = (l == 0) ? E_DIM : 2 * H_;
    int Kp = (l == 0) ? KE : K2H;
    pack(wEncFt[l], encWihF[l], H3, K, N3H, Kp);
    pack(wEncBt[l], encWihB[l], H3, K, N3H, Kp);
  }
  pack(linQt,   INF(43), H_, 2 * H_, NH, K2H);   // lin_question
  pack(linPt,   INF(41), H_, 2 * H_, NH, K2H);   // lin_passage
  pack(linSPt,  INF(45), H_, 2 * H_, NH, K2H);   // lin_self_p
  pack(linSPWt, INF(47), H_, 2 * H_, NH, K2H);   // lin_self_pw
  pack(linHPt,  INF(39), H_, 2 * H_, NH, K2H);   // lin_hp
  pack(vtPk,    INF(49), H_, H_,     NH, KH);    // lin_vt

  auto gemm = [&](const h16* A, int lda, const h16* Bpk, const float* bias,
                  float* C, int ldc, int M, int Nvalid, int Kp, int Np) {
    int Mt = (M + 15) / 16, Nt = Np / 16;
    int tiles = Mt * Nt;
    wmma_gemm_bias<<<(tiles + 3) / 4, 128, 0, stream>>>(A, lda, Bpk, Nt, bias,
                                                        C, ldc, M, Nvalid, Kp, Mt);
  };

  // ---- encoder (passage then question) ----
  auto run_encoder = [&](const int* tok, int T, h16* x0h, float* uF, h16* uFh, float* yScr) {
    gather_emb_kernel<<<(T * KE + 255) / 256, 256, 0, stream>>>(x0h, tok, emb, T);
    const h16* As[3] = { x0h, x1h, x2h };
    int ldas[3] = { KE, K2H, K2H };
    for (int l = 0; l < 3; ++l) {
      gemm(As[l], ldas[l], wEncFt[l], encBihF[l], giF, N3H, T, H3, ldas[l], N3H);
      gemm(As[l], ldas[l], wEncBt[l], encBihB[l], giB, N3H, T, H3, ldas[l], N3H);
      float* y = (l == 2) ? uF : yScr;
      h16* yh = (l == 0) ? x1h : (l == 1) ? x2h : uFh;
      gru_scan_kernel<<<1, 256, 0, stream>>>(giF, N3H, encWhhF[l], encBhhF[l],
                                             y, yh, K2H, 0, T, 0);
      gru_scan_kernel<<<1, 256, 0, stream>>>(giB, N3H, encWhhB[l], encBhhB[l],
                                             y, yh, K2H, H_, T, 1);
    }
  };
  run_encoder(passage,  P_LEN, xPh, uP, uPh, uP);
  run_encoder(question, Q_LEN, xQh, uQ, uQh, uQ);

  // qproj = lin_question(u_q) (ld 96); r_q = attention pool
  gemm(uQh, K2H, linQt, INF(44), qproj, KH, Q_LEN, H_, K2H, NH);
  pool_question_kernel<<<1, 96, 0, stream>>>(qproj, INF(49), INF(50), rq);

  // ppr_all = lin_passage(u_p)
  gemm(uPh, K2H, linPt, INF(42), pprA, NH, P_LEN, H_, K2H, NH);

  // question-aware passage scan
  qa_scan_kernel<<<1, 512, 0, stream>>>(
      qproj, pprA, uP, vtPk,
      INF(37), INF(38),            // lin_hidden W,b
      INF(31), INF(32),            // gate_qa W,b
      INF(58), INF(57), INF(60), INF(59),   // qa_rnn.f Wih,Whh,bih,bhh
      INF(54), INF(53), INF(56), INF(55),   // qa_rnn.b
      vbuf, vh, P_LEN);

  // self-matching scan
  gemm(vh, K2H, linSPt,  INF(46), pproj, KH, P_LEN, H_, K2H, NH);   // ld 96
  gemm(vh, K2H, linSPWt, INF(48), pwrA,  NH, P_LEN, H_, K2H, NH);
  sm_scan_kernel<<<1, 512, 0, stream>>>(
      pproj, pwrA, vbuf, vtPk,
      INF(33), INF(34),            // gate_sm W,b
      INF(66), INF(65), INF(68), INF(67),   // sm_rnn.f
      INF(62), INF(61), INF(64), INF(63),   // sm_rnn.b
      hsm, hsmh, P_LEN);

  // pointer network
  gemm(hsmh, K2H, linHPt, INF(40), hp, NH, P_LEN, H_, K2H, NH);
  pointer_kernel<<<1, 256, 0, stream>>>(
      hp, rq,
      INF(35), INF(36),            // lin_ha
      INF(51), INF(52),            // lin_vt_ans
      INF(3), INF(2), INF(5), INF(4),       // ans_rnn Wih,Whh,bih,bhh
      (float*)d_out);
#undef INF
}
